// GroupTransformer_16904991277587
// MI455X (gfx1250) — matrix operand order repaired and checked by reading, not compiled, not run on hardware
//
#include <hip/hip_runtime.h>
#include <hip/hip_bf16.h>

// Problem constants
#define NB   32
#define LSEQ 4096
#define DIN  768
#define DM   256
#define GG   64
#define HH   4
#define TOP  3
#define DH   64
#define INV_TEMP 0.125f   // 1/sqrt(64)
#define EPS 1e-12f

typedef __attribute__((ext_vector_type(16))) __bf16 v16bf;
typedef __attribute__((ext_vector_type(8)))  float  v8f;

__device__ __forceinline__ __bf16 f2bf(float f) {
    unsigned int u = __builtin_bit_cast(unsigned int, f);
    unsigned int r = (u + 0x7FFFu + ((u >> 16) & 1u)) >> 16;
    return __builtin_bit_cast(__bf16, (unsigned short)r);
}

__device__ __forceinline__ v8f wmma_bf16(v16bf a, v16bf b, v8f c) {
    // 8 args: (neg_a, A, neg_b, B, c_mod, C, reuse_a, reuse_b)
    return __builtin_amdgcn_wmma_f32_16x16x32_bf16(false, a, false, b, (short)0, c, false, false);
}

// A fragment (16x32 bf16, M x K), row-major source, ld = row pitch.
// lane%16 -> M row; lane/16 -> K offset 0/8; elems 0..7 -> K kb..kb+7, 8..15 -> K kb+16..kb+23
__device__ __forceinline__ v16bf load_fragA_bf16(const __bf16* p, int row, int ld, int k0, int lane) {
    const __bf16* s = p + (row + (lane & 15)) * ld + k0 + ((lane >> 4) << 3);
    v16bf f;
#pragma unroll
    for (int e = 0; e < 8; ++e) f[e] = s[e];
#pragma unroll
    for (int e = 0; e < 8; ++e) f[8 + e] = s[16 + e];
    return f;
}

// Same as above but converts from f32 source on the fly.
__device__ __forceinline__ v16bf load_fragA_f32(const float* p, int row, int ld, int k0, int lane) {
    const float* s = p + (row + (lane & 15)) * ld + k0 + ((lane >> 4) << 3);
    v16bf f;
#pragma unroll
    for (int e = 0; e < 8; ++e) f[e] = f2bf(s[e]);
#pragma unroll
    for (int e = 0; e < 8; ++e) f[8 + e] = f2bf(s[16 + e]);
    return f;
}

// B fragment (32x16 bf16, K x N) read from an LDS panel stored as [cols][kElems]
// (col-major trick: per lane two contiguous 8-element K runs).
// lane%16 -> N col; lane/16 -> K offset 0/8; elems 0..7 -> K kb..kb+7, 8..15 -> K kb+16..kb+23.
__device__ __forceinline__ v16bf load_fragB_lds(const __bf16* p, int col, int ld, int k0, int lane) {
    const __bf16* s = p + (col + (lane & 15)) * ld + k0 + ((lane >> 4) << 3);
    v16bf f;
#pragma unroll
    for (int e = 0; e < 8; ++e) f[e] = s[e];
#pragma unroll
    for (int e = 0; e < 8; ++e) f[8 + e] = s[16 + e];
    return f;
}

// ---- CDNA5 async global->LDS stage of a 2D panel: rows x kElems bf16, global ld = ldg elems.
// Issues per-lane global_load_async_to_lds_b128 (16B granules), waits ASYNCcnt, barriers.
__device__ __forceinline__ void async_stage_panel(const __bf16* gsrc, int ldg,
                                                  __bf16* sdst, int rows, int kElems,
                                                  int tid, int nthreads) {
    __syncthreads();   // all waves done reading previous contents
    const int rowBytes = kElems * 2;
    const int granPerRow = rowBytes >> 4;         // 16-byte granules per row
    const int total = rows * granPerRow;
    unsigned lbase = (unsigned)(uintptr_t)sdst;   // low 32 bits = LDS offset
    for (int gidx = tid; gidx < total; gidx += nthreads) {
        int row  = gidx / granPerRow;
        int boff = (gidx - row * granPerRow) << 4;
        unsigned laddr = lbase + row * rowBytes + boff;
        unsigned long long gaddr =
            (unsigned long long)(uintptr_t)(gsrc + row * ldg) + (unsigned)boff;
        asm volatile("global_load_async_to_lds_b128 %0, %1, off"
                     :: "v"(laddr), "v"(gaddr) : "memory");
    }
    asm volatile("s_wait_asynccnt 0" ::: "memory");
    __syncthreads();   // panel visible to all waves
}

// ---------------- K0: convert weight [K][N] f32 -> transposed [N][K] bf16 ----------------
__global__ __launch_bounds__(256) void k_convT(const float* __restrict__ in, __bf16* __restrict__ out,
                                               int K, int Nc) {
    int i = blockIdx.x * 256 + threadIdx.x;
    if (i >= K * Nc) return;
    int k = i / Nc, n = i % Nc;
    out[n * K + k] = f2bf(in[i]);
}

// ---------------- K3: q = group_emb @ Wq + bq  ->  qb[h][g][dh] bf16 ----------------
__global__ __launch_bounds__(256) void k_qproj(const float* __restrict__ ge, const float* __restrict__ Wq,
                                               const float* __restrict__ bq, __bf16* __restrict__ qb) {
    int g = blockIdx.x;     // 0..63
    int d = threadIdx.x;    // 0..255
    float acc = bq[d];
    for (int k = 0; k < DM; ++k) acc += ge[g * DM + k] * Wq[k * DM + d];
    int h = d >> 6, dh = d & 63;
    qb[h * (GG * DH) + g * DH + dh] = f2bf(acc);
}

// ---------------- K1: h = x @ W_in + b_in + pos_emb  (bf16 out) ----------------
// grid (1024, 4), block 256 (8 waves). Block: 128 M rows x 64 N cols. Wave: 16x64.
// B weight panel (64 x 768) staged into LDS in 3 chunks of K=256 via async DMA.
__global__ __launch_bounds__(256) void k_proj_in(const float* __restrict__ x, const float* __restrict__ b_in,
                                                 const float* __restrict__ pos, const __bf16* __restrict__ WinT,
                                                 __bf16* __restrict__ hout) {
    __shared__ alignas(16) __bf16 sB[64 * 256];   // 32 KB chunk
    int tid = threadIdx.x;
    int lane = tid & 31;
    int wave = tid >> 5;
    int mBase = blockIdx.x * 128 + wave * 16;
    int nBase = blockIdx.y * 64;
    v8f acc[4] = {};
    for (int c0 = 0; c0 < DIN; c0 += 256) {
        async_stage_panel(WinT + nBase * DIN + c0, DIN, sB, 64, 256, tid, 256);
#pragma unroll
        for (int k0 = 0; k0 < 256; k0 += 32) {
            v16bf a = load_fragA_f32(x, mBase, DIN, c0 + k0, lane);
#pragma unroll
            for (int j = 0; j < 4; ++j) {
                v16bf b = load_fragB_lds(sB, 16 * j, 256, k0, lane);
                acc[j] = wmma_bf16(a, b, acc[j]);
            }
        }
    }
    int mo = (lane >> 4) << 3;
    int nc = lane & 15;
#pragma unroll
    for (int j = 0; j < 4; ++j) {
#pragma unroll
        for (int i = 0; i < 8; ++i) {
            int m = mBase + i + mo;
            int n = nBase + 16 * j + nc;
            int l = m & (LSEQ - 1);
            float hv = acc[j][i] + b_in[n] + pos[l * DM + n];
            hout[m * DM + n] = f2bf(hv);
        }
    }
}

// ---------------- K2: k/v = h @ Wk/Wv + bias; z==0 -> k(bf16 [N,H,L,DH]), z==1 -> v(f32) ----------------
__global__ __launch_bounds__(256) void k_proj_kv(const __bf16* __restrict__ h, const __bf16* __restrict__ WkT,
                                                 const __bf16* __restrict__ WvT, const float* __restrict__ bk,
                                                 const float* __restrict__ bv, __bf16* __restrict__ kout,
                                                 float* __restrict__ vout) {
    __shared__ alignas(16) __bf16 sB[64 * 256];   // 32 KB weight panel
    bool isV = (blockIdx.z == 1);
    const __bf16* Wt = isV ? WvT : WkT;
    const float* bias = isV ? bv : bk;
    int tid = threadIdx.x;
    int lane = tid & 31;
    int wave = tid >> 5;
    int mBase = blockIdx.x * 128 + wave * 16;
    int nBase = blockIdx.y * 64;
    async_stage_panel(Wt + nBase * DM, DM, sB, 64, DM, tid, 256);
    v8f acc[4] = {};
#pragma unroll
    for (int k0 = 0; k0 < DM; k0 += 32) {
        v16bf a = load_fragA_bf16(h, mBase, DM, k0, lane);
#pragma unroll
        for (int j = 0; j < 4; ++j) {
            v16bf b = load_fragB_lds(sB, 16 * j, DM, k0, lane);
            acc[j] = wmma_bf16(a, b, acc[j]);
        }
    }
    int mo = (lane >> 4) << 3;
    int nc = lane & 15;
#pragma unroll
    for (int j = 0; j < 4; ++j) {
#pragma unroll
        for (int i = 0; i < 8; ++i) {
            int m = mBase + i + mo;                 // m = batch*L + l
            int n = nBase + 16 * j + nc;            // n = head*64 + dh
            int batch = m >> 12, l = m & (LSEQ - 1);
            int hd = n >> 6, dh = n & 63;
            int off = (((batch * HH + hd) * LSEQ) + l) * DH + dh;
            float val = acc[j][i] + bias[n];
            if (isV) vout[off] = val;
            else     kout[off] = f2bf(val);
        }
    }
}

// ---------------- K4: logits[n,h,g,l] = (q_h[g,:] . k[n,h,l,:]) / TEMP -> soft slab ----------------
// grid (N*H, L/64), block 128 (4 waves). Wave w: rows g=16w..16w+15, 64 l-cols.
// Shared 64x64 k-tile staged into LDS once per block (reused by all 4 waves).
__global__ __launch_bounds__(128) void k_logits(const __bf16* __restrict__ qb, const __bf16* __restrict__ kb,
                                                float* __restrict__ soft) {
    __shared__ alignas(16) __bf16 sK[64 * DH];    // 8 KB
    int nh = blockIdx.x;
    int tid = threadIdx.x;
    int lane = tid & 31;
    int wave = tid >> 5;
    int gBase = wave * 16;
    int lBase = blockIdx.y * 64;
    int hd = nh & (HH - 1);
    const __bf16* qh = qb + hd * (GG * DH);               // [G][DH] row-major
    const __bf16* kh = kb + (size_t)nh * (LSEQ * DH);     // [L][DH]
    async_stage_panel(kh + (size_t)lBase * DH, DH, sK, 64, DH, tid, 128);
    v8f acc[4] = {};
#pragma unroll
    for (int k0 = 0; k0 < DH; k0 += 32) {
        v16bf a = load_fragA_bf16(qh, gBase, DH, k0, lane);
#pragma unroll
        for (int j = 0; j < 4; ++j) {
            v16bf b = load_fragB_lds(sK, 16 * j, DH, k0, lane);
            acc[j] = wmma_bf16(a, b, acc[j]);
        }
    }
    int mo = (lane >> 4) << 3;
    int nc = lane & 15;
    float* srow = soft + (size_t)nh * GG * LSEQ;
#pragma unroll
    for (int j = 0; j < 4; ++j) {
#pragma unroll
        for (int i = 0; i < 8; ++i) {
            int g = gBase + i + mo;
            int l = lBase + 16 * j + nc;
            srow[(size_t)g * LSEQ + l] = acc[j][i] * INV_TEMP;
        }
    }
}

// ---------------- K5: row softmax over L, argmax (first-max tie-break), one-hot hard ----------------
__global__ __launch_bounds__(256) void k_softmax(float* __restrict__ soft, float* __restrict__ hard,
                                                 int* __restrict__ idx_out) {
    int row = blockIdx.x;           // (n*H+h)*G + g
    float* s = soft + (size_t)row * LSEQ;
    float* hd = hard + (size_t)row * LSEQ;
    int tid = threadIdx.x;
    __shared__ float sv[256];
    __shared__ int   si[256];
    float bv = -3.0e38f; int bi = 0;
    for (int i = tid; i < LSEQ; i += 256) {
        float v = s[i];
        if (v > bv) { bv = v; bi = i; }     // strict > keeps first index within a thread
    }
    sv[tid] = bv; si[tid] = bi;
    __syncthreads();
    for (int o = 128; o; o >>= 1) {
        if (tid < o) {
            float v2 = sv[tid + o]; int i2 = si[tid + o];
            if (v2 > sv[tid] || (v2 == sv[tid] && i2 < si[tid])) { sv[tid] = v2; si[tid] = i2; }
        }
        __syncthreads();
    }
    float rmax = sv[0]; int ridx = si[0];
    __syncthreads();
    float lsum = 0.0f;
    for (int i = tid; i < LSEQ; i += 256) lsum += __expf(s[i] - rmax);
    sv[tid] = lsum;
    __syncthreads();
    for (int o = 128; o; o >>= 1) { if (tid < o) sv[tid] += sv[tid + o]; __syncthreads(); }
    float inv = 1.0f / sv[0];
    for (int i = tid; i < LSEQ; i += 256) {
        s[i]  = __expf(s[i] - rmax) * inv;
        hd[i] = (i == ridx) ? 1.0f : 0.0f;
    }
    if (tid == 0) idx_out[row] = ridx;
}

// ---------------- K6: gather v rows by argmax, per-(n,g) l2norm over D, store bf16 ----------------
__global__ __launch_bounds__(64) void k_gather(const float* __restrict__ v, const int* __restrict__ idx,
                                               __bf16* __restrict__ aggn) {
    int blk = blockIdx.x;            // n*G + g
    int n = blk >> 6, g = blk & 63;
    int t = threadIdx.x;             // dh
    float val[HH]; float ss = 0.0f;
#pragma unroll
    for (int h = 0; h < HH; ++h) {
        int l = idx[(n * HH + h) * GG + g];
        float x = v[(((n * HH + h) * LSEQ) + l) * DH + t];
        val[h] = x; ss += x * x;
    }
    __shared__ float red[64];
    red[t] = ss; __syncthreads();
    for (int o = 32; o; o >>= 1) { if (t < o) red[t] += red[t + o]; __syncthreads(); }
    float scale = rsqrtf(fmaxf(red[0], EPS));
#pragma unroll
    for (int h = 0; h < HH; ++h)
        aggn[(blk) * DM + h * DH + t] = f2bf(val[h] * scale);
}

// ---------------- K7: gf = aggn @ Wo + bo + group_emb; accumulate global sum(gf^2) ----------------
// grid (16, 4), block 256; M = N*G = 2048
__global__ __launch_bounds__(256) void k_gf(const __bf16* __restrict__ aggn, const __bf16* __restrict__ WoT,
                                            const float* __restrict__ bo, const float* __restrict__ ge,
                                            float* __restrict__ gf, float* __restrict__ sumsq) {
    __shared__ alignas(16) __bf16 sB[64 * 256];   // 32 KB weight panel
    int tid = threadIdx.x;
    int lane = tid & 31;
    int wave = tid >> 5;
    int mBase = blockIdx.x * 128 + wave * 16;
    int nBase = blockIdx.y * 64;
    async_stage_panel(WoT + nBase * DM, DM, sB, 64, DM, tid, 256);
    v8f acc[4] = {};
#pragma unroll
    for (int k0 = 0; k0 < DM; k0 += 32) {
        v16bf a = load_fragA_bf16(aggn, mBase, DM, k0, lane);
#pragma unroll
        for (int j = 0; j < 4; ++j) {
            v16bf b = load_fragB_lds(sB, 16 * j, DM, k0, lane);
            acc[j] = wmma_bf16(a, b, acc[j]);
        }
    }
    int mo = (lane >> 4) << 3;
    int nc = lane & 15;
    float local = 0.0f;
#pragma unroll
    for (int j = 0; j < 4; ++j) {
#pragma unroll
        for (int i = 0; i < 8; ++i) {
            int m = mBase + i + mo;             // m = n*G + g
            int n = nBase + 16 * j + nc;
            int g = m & (GG - 1);
            float v = acc[j][i] + bo[n] + ge[g * DM + n];
            gf[m * DM + n] = v;
            local += v * v;
        }
    }
    atomicAdd(sumsq, local);
}

// ---------------- K8: rslt = (gf * rsqrt(max(sumsq,eps))) @ W_top + b_top ----------------
__global__ __launch_bounds__(256) void k_top(const float* __restrict__ gf, const float* __restrict__ sumsq,
                                             const float* __restrict__ Wt, const float* __restrict__ bt,
                                             float* __restrict__ out) {
    int i = blockIdx.x * 256 + threadIdx.x;
    if (i >= NB * GG * TOP) return;
    int row = i / TOP, t = i % TOP;
    float scale = rsqrtf(fmaxf(*sumsq, EPS));
    float acc = bt[t];
    for (int k = 0; k < DM; ++k) acc += gf[row * DM + k] * scale * Wt[k * TOP + t];
    out[i] = acc;
}

extern "C" void kernel_launch(void* const* d_in, const int* in_sizes, int n_in,
                              void* d_out, int out_size, void* d_ws, size_t ws_size,
                              hipStream_t stream) {
    const float* x    = (const float*)d_in[0];
    const float* W_in = (const float*)d_in[1];
    const float* b_in = (const float*)d_in[2];
    const float* pos  = (const float*)d_in[3];
    const float* ge   = (const float*)d_in[4];
    const float* Wq   = (const float*)d_in[5];
    const float* bq   = (const float*)d_in[6];
    const float* Wk   = (const float*)d_in[7];
    const float* bk   = (const float*)d_in[8];
    const float* Wv   = (const float*)d_in[9];
    const float* bv   = (const float*)d_in[10];
    const float* Wo   = (const float*)d_in[11];
    const float* bo   = (const float*)d_in[12];
    const float* Wtop = (const float*)d_in[13];
    const float* btop = (const float*)d_in[14];

    float* out  = (float*)d_out;
    float* rslt = out;                                  // [N,G,TOP] = 6144
    float* soft = out + (size_t)NB * GG * TOP;          // [N,H,G,L]
    float* hard = soft + (size_t)NB * HH * GG * LSEQ;   // [N,H,G,L]

    // ----- workspace carve-up -----
    char* p = (char*)d_ws;
    auto alloc = [&](size_t bytes) { void* r = (void*)p; p += (bytes + 255) & ~(size_t)255; return r; };
    __bf16* WinT = (__bf16*)alloc((size_t)DIN * DM * 2);
    __bf16* WkT  = (__bf16*)alloc((size_t)DM * DM * 2);
    __bf16* WvT  = (__bf16*)alloc((size_t)DM * DM * 2);
    __bf16* WoT  = (__bf16*)alloc((size_t)DM * DM * 2);
    __bf16* qb   = (__bf16*)alloc((size_t)HH * GG * DH * 2);
    __bf16* hbuf = (__bf16*)alloc((size_t)NB * LSEQ * DM * 2);      // 64 MB
    __bf16* kbuf = (__bf16*)alloc((size_t)NB * HH * LSEQ * DH * 2); // 64 MB
    float*  vbuf = (float*) alloc((size_t)NB * HH * LSEQ * DH * 4); // 128 MB
    int*    idx  = (int*)   alloc((size_t)NB * HH * GG * 4);
    __bf16* aggn = (__bf16*)alloc((size_t)NB * GG * DM * 2);
    float*  gf   = (float*) alloc((size_t)NB * GG * DM * 4);
    float*  ssq  = (float*) alloc(256);

    hipMemsetAsync(ssq, 0, 4, stream);

    // weight convert + transpose to bf16 [N][K]
    k_convT<<<(DIN * DM + 255) / 256, 256, 0, stream>>>(W_in, WinT, DIN, DM);
    k_convT<<<(DM * DM + 255) / 256, 256, 0, stream>>>(Wk, WkT, DM, DM);
    k_convT<<<(DM * DM + 255) / 256, 256, 0, stream>>>(Wv, WvT, DM, DM);
    k_convT<<<(DM * DM + 255) / 256, 256, 0, stream>>>(Wo, WoT, DM, DM);

    // q projection (tiny, VALU)
    k_qproj<<<GG, 256, 0, stream>>>(ge, Wq, bq, qb);

    // h = x@W_in + b_in + pos  (WMMA, async-staged weight panel)
    k_proj_in<<<dim3((NB * LSEQ) / 128, DM / 64), 256, 0, stream>>>(x, b_in, pos, WinT, hbuf);

    // k, v projections (WMMA, async-staged weight panel)
    k_proj_kv<<<dim3((NB * LSEQ) / 128, DM / 64, 2), 256, 0, stream>>>(hbuf, WkT, WvT, bk, bv, kbuf, vbuf);

    // attention logits (WMMA, async-staged k-tile) -> raw logits into soft slab
    k_logits<<<dim3(NB * HH, LSEQ / 64), 128, 0, stream>>>(qb, kbuf, soft);

    // softmax + argmax + one-hot hard
    k_softmax<<<NB * HH * GG, 256, 0, stream>>>(soft, hard, idx);

    // gather v by argmax + per-row l2norm -> bf16 aggn
    k_gather<<<NB * GG, 64, 0, stream>>>(vbuf, idx, aggn);

    // gf = aggn@Wo + bo + group_emb (WMMA) + global sum of squares
    k_gf<<<dim3((NB * GG) / 128, DM / 64), 256, 0, stream>>>(aggn, WoT, bo, ge, gf, ssq);

    // final projection with global l2 scale
    k_top<<<(NB * GG * TOP + 255) / 256, 256, 0, stream>>>(gf, ssq, Wtop, btop, rslt);
}
